// GNNEncoder_7481833029725
// MI455X (gfx1250) — compile-verified, matrix operand order checked by reading, not run on hardware
//
#include <hip/hip_runtime.h>
#include <hip/hip_bf16.h>

// ---------------- problem constants (match reference) ----------------
#define NNODES 100000
#define NEDGES 1600000
#define CIN    256
#define CH     128
#define COUT   64
#define NB     64
#define BN_EPS 1e-5f

typedef __bf16 bf16_t;
typedef __attribute__((ext_vector_type(16))) __bf16 v16bf;
typedef __attribute__((ext_vector_type(8)))  __bf16 v8bf;
typedef __attribute__((ext_vector_type(8)))  float  v8f;

// ---------------- small utility kernels ----------------
__global__ void fill_f32(float* __restrict__ p, float v, int n) {
    int i = blockIdx.x * blockDim.x + threadIdx.x;
    if (i < n) p[i] = v;
}

__global__ void deg_count(const int* __restrict__ dst, float* __restrict__ deg, int e) {
    int i = blockIdx.x * blockDim.x + threadIdx.x;
    if (i < e) atomicAdd(&deg[dst[i]], 1.0f);
}

__global__ void rsqrt_inplace(float* __restrict__ p, int n) {
    int i = blockIdx.x * blockDim.x + threadIdx.x;
    if (i < n) p[i] = rsqrtf(p[i]);
}

__global__ void cast_f32_bf16(const float* __restrict__ x, bf16_t* __restrict__ y, int n) {
    int i = blockIdx.x * blockDim.x + threadIdx.x;
    if (i < n) y[i] = (bf16_t)x[i];
}

// Repack a row-major [K, Hout] fp32 weight into the WMMA B-operand layout
// (16x16x32 bf16): per (kt, nt) tile, 32 lanes x 16 contiguous bf16 elements.
// Lane l (half = l>>4, n = l&15):
//   e in [0,8):  k = kt*32 + half*8 + e
//   e in [8,16): k = kt*32 + 16 + half*8 + (e-8)
__global__ void pack_weight_b(const float* __restrict__ W, bf16_t* __restrict__ Wp,
                              int K, int Hout) {
    int t = blockIdx.x * blockDim.x + threadIdx.x;
    int total = K * Hout;
    if (t >= total) return;
    int e    = t & 15;
    int lane = (t >> 4) & 31;
    int tile = t >> 9;
    int ntn  = Hout >> 4;
    int nt   = tile % ntn;
    int kt   = tile / ntn;
    int half = lane >> 4;
    int n    = (lane & 15) + nt * 16;
    int kin  = (e < 8) ? (half * 8 + e) : (16 + half * 8 + (e - 8));
    int k    = kt * 32 + kin;
    Wp[t] = (bf16_t)W[(size_t)k * Hout + n];
}

// ---------------- WMMA GEMM: Y[nrows,HOUT] = A[nrows,K] (bf16) x Wp (packed bf16) -------------
// One wave per 16-row slab; ALL HOUT/16 column tiles accumulated in registers so the
// A tile is loaded once per kt and feeds 8 independent WMMA chains (no RAW hazard).
template<int HOUT>
__global__ __launch_bounds__(32)
void gemm_bf16_wmma(const bf16_t* __restrict__ A, const bf16_t* __restrict__ Wp,
                    const float* __restrict__ bias, float* __restrict__ Y,
                    int nrows, int K) {
    constexpr int NTN = HOUT / 16;
    const int rowBase = blockIdx.x * 16;
    if (rowBase >= nrows) return;
    const int lane = threadIdx.x;
    const int half = lane >> 4;
    const int am   = lane & 15;
    const int kTiles = K >> 5;

    const bf16_t* arow = A + (size_t)(rowBase + am) * K + half * 8;

    v8f c[NTN];
#pragma unroll
    for (int t = 0; t < NTN; ++t) c[t] = (v8f){};

    for (int kt = 0; kt < kTiles; ++kt) {
        // A operand: ISA 16-bit A layout -> two 16-byte chunks per lane
        v8bf a0 = *(const v8bf*)(arow + kt * 32);
        v8bf a1 = *(const v8bf*)(arow + kt * 32 + 16);
        v16bf a;
#pragma unroll
        for (int i = 0; i < 8; ++i) { a[i] = a0[i]; a[i + 8] = a1[i]; }
        // B operand: pre-packed, 16 contiguous bf16 per lane per tile
        const bf16_t* wbase = Wp + ((size_t)kt * NTN * 32 + lane) * 16;
#pragma unroll
        for (int t = 0; t < NTN; ++t) {
            v16bf b = *(const v16bf*)(wbase + (size_t)t * 512);
            c[t] = __builtin_amdgcn_wmma_f32_16x16x32_bf16(false, a, false, b,
                                                           (short)0, c[t], false, false);
        }
    }

    const int ncol = lane & 15;
#pragma unroll
    for (int t = 0; t < NTN; ++t) {
        const int col = t * 16 + ncol;
        const float badd = bias ? bias[col] : 0.0f;
#pragma unroll
        for (int v = 0; v < 8; ++v) {
            // C/D layout: lanes 16-31 hold M+8
            Y[(size_t)(rowBase + v + half * 8) * HOUT + col] = c[t][v] + badd;
        }
    }
}

// ---------------- edge aggregation: one wave per edge, 4 ch/lane (H=128) ----------------
__global__ __launch_bounds__(256)
void edge_aggregate(const float* __restrict__ hlin, const int* __restrict__ src,
                    const int* __restrict__ dst, const float* __restrict__ dinv,
                    float* __restrict__ agg, int e) {
    int gid  = blockIdx.x * blockDim.x + threadIdx.x;
    int edge = gid >> 5;
    int lane = gid & 31;
    if (edge >= e) return;
    int s = src[edge], d = dst[edge];
    float en = dinv[s] * dinv[d];
    float4 v = *(const float4*)(hlin + (size_t)s * CH + lane * 4);
    float* ap = agg + (size_t)d * CH + lane * 4;
    atomicAdd(ap + 0, v.x * en);
    atomicAdd(ap + 1, v.y * en);
    atomicAdd(ap + 2, v.z * en);
    atomicAdd(ap + 3, v.w * en);
}

// ---------------- fused epilogue: agg + self-loop + bias -> BN -> ReLU (+res) -> f32 & bf16 ----
template<bool HAS_RES>
__global__ void combine_bn_relu(const float* __restrict__ agg, const float* __restrict__ hlin,
                                const float* __restrict__ dinv, const float* __restrict__ b,
                                const float* __restrict__ g, const float* __restrict__ be,
                                const float* __restrict__ res, const float* __restrict__ bres,
                                float* __restrict__ h, bf16_t* __restrict__ hh, int n) {
    int i = blockIdx.x * blockDim.x + threadIdx.x;
    if (i >= n) return;
    int node = i >> 7;               // / CH
    int c    = i & 127;              // % CH
    float di = dinv[node];
    float v  = agg[i] + hlin[i] * di * di + b[c];
    v = v * (g[c] * rsqrtf(1.0f + BN_EPS)) + be[c];
    v = fmaxf(v, 0.0f);
    if (HAS_RES) v += res[i] + bres[c];   // residual added after ReLU (reference semantics)
    h[i]  = v;
    hh[i] = (bf16_t)v;
}

__global__ void relu_bias_cast(const float* __restrict__ t, const float* __restrict__ b,
                               bf16_t* __restrict__ y, int n) {
    int i = blockIdx.x * blockDim.x + threadIdx.x;
    if (i < n) y[i] = (bf16_t)fmaxf(t[i] + b[i & 127], 0.0f);
}

// ---------------- global mean pool ----------------
__global__ void count_nodes(const int* __restrict__ batch, float* __restrict__ counts, int n) {
    int i = blockIdx.x * blockDim.x + threadIdx.x;
    if (i < n) atomicAdd(&counts[batch[i]], 1.0f);
}

__global__ __launch_bounds__(256)
void pool_sum(const float* __restrict__ h, const int* __restrict__ batch,
              float* __restrict__ pooled, int n) {
    int gid  = blockIdx.x * blockDim.x + threadIdx.x;
    int node = gid >> 5;
    int lane = gid & 31;
    if (node >= n) return;
    int b = batch[node];
    float4 v = *(const float4*)(h + (size_t)node * CH + lane * 4);
    float* pp = pooled + (size_t)b * CH + lane * 4;
    atomicAdd(pp + 0, v.x);
    atomicAdd(pp + 1, v.y);
    atomicAdd(pp + 2, v.z);
    atomicAdd(pp + 3, v.w);
}

__global__ void pool_div(float* __restrict__ pooled, const float* __restrict__ counts, int n) {
    int i = blockIdx.x * blockDim.x + threadIdx.x;
    if (i < n) pooled[i] /= fmaxf(counts[i >> 7], 1.0f);
}

// ---------------- host-side orchestration ----------------
static inline int cdiv(int a, int b) { return (a + b - 1) / b; }

extern "C" void kernel_launch(void* const* d_in, const int* in_sizes, int n_in,
                              void* d_out, int out_size, void* d_ws, size_t ws_size,
                              hipStream_t stream) {
    (void)in_sizes; (void)n_in; (void)out_size; (void)ws_size;

    const float* x     = (const float*)d_in[0];
    const int*   src   = (const int*)d_in[1];            // edge_index[0]
    const int*   dst   = ((const int*)d_in[1]) + NEDGES; // edge_index[1]
    const int*   batch = (const int*)d_in[2];
    const float* W0   = (const float*)d_in[3];  const float* b0  = (const float*)d_in[4];
    const float* W1   = (const float*)d_in[5];  const float* b1  = (const float*)d_in[6];
    const float* W2   = (const float*)d_in[7];  const float* b2  = (const float*)d_in[8];
    const float* g0   = (const float*)d_in[9];  const float* be0 = (const float*)d_in[10];
    const float* g1   = (const float*)d_in[11]; const float* be1 = (const float*)d_in[12];
    const float* g2   = (const float*)d_in[13]; const float* be2 = (const float*)d_in[14];
    const float* Wres = (const float*)d_in[15]; const float* bres = (const float*)d_in[16];
    const float* Wl0  = (const float*)d_in[17]; const float* bl0  = (const float*)d_in[18];
    const float* Wl1  = (const float*)d_in[19]; const float* bl1  = (const float*)d_in[20];

    float* out    = (float*)d_out;                           // [N, COUT]
    float* pooled = ((float*)d_out) + (size_t)NNODES * COUT; // [NB, CH]

    // carve workspace
    char* ws = (char*)d_ws;
    size_t off = 0;
    auto carve = [&](size_t bytes) { char* p = ws + off; off = (off + bytes + 255) & ~(size_t)255; return p; };
    bf16_t* xh    = (bf16_t*)carve((size_t)NNODES * CIN * 2);
    bf16_t* hh    = (bf16_t*)carve((size_t)NNODES * CH  * 2);
    float*  hlin  = (float*) carve((size_t)NNODES * CH  * 4);
    float*  agg   = (float*) carve((size_t)NNODES * CH  * 4);
    float*  h     = (float*) carve((size_t)NNODES * CH  * 4);
    float*  res   = (float*) carve((size_t)NNODES * CH  * 4);
    float*  dinv  = (float*) carve((size_t)NNODES * 4);
    float*  cnts  = (float*) carve((size_t)NB * 4);
    bf16_t* W0p   = (bf16_t*)carve((size_t)CIN * CH * 2);
    bf16_t* Wresp = (bf16_t*)carve((size_t)CIN * CH * 2);
    bf16_t* W1p   = (bf16_t*)carve((size_t)CH * CH * 2);
    bf16_t* W2p   = (bf16_t*)carve((size_t)CH * CH * 2);
    bf16_t* Wl0p  = (bf16_t*)carve((size_t)CH * CH * 2);
    bf16_t* Wl1p  = (bf16_t*)carve((size_t)CH * COUT * 2);

    const int NH = NNODES * CH;
    const dim3 blk256(256);
    const dim3 gemmGrid(NNODES / 16);
    const dim3 wblk(32);

    // degrees (self loop => init 1.0), then dinv = rsqrt(deg)
    fill_f32<<<cdiv(NNODES, 256), blk256, 0, stream>>>(dinv, 1.0f, NNODES);
    deg_count<<<cdiv(NEDGES, 256), blk256, 0, stream>>>(dst, dinv, NEDGES);
    rsqrt_inplace<<<cdiv(NNODES, 256), blk256, 0, stream>>>(dinv, NNODES);

    // casts + weight repacks
    cast_f32_bf16<<<cdiv(NNODES * CIN, 256), blk256, 0, stream>>>(x, xh, NNODES * CIN);
    pack_weight_b<<<cdiv(CIN * CH, 256), blk256, 0, stream>>>(W0,   W0p,   CIN, CH);
    pack_weight_b<<<cdiv(CIN * CH, 256), blk256, 0, stream>>>(Wres, Wresp, CIN, CH);
    pack_weight_b<<<cdiv(CH * CH, 256),  blk256, 0, stream>>>(W1,   W1p,   CH, CH);
    pack_weight_b<<<cdiv(CH * CH, 256),  blk256, 0, stream>>>(W2,   W2p,   CH, CH);
    pack_weight_b<<<cdiv(CH * CH, 256),  blk256, 0, stream>>>(Wl0,  Wl0p,  CH, CH);
    pack_weight_b<<<cdiv(CH * COUT, 256),blk256, 0, stream>>>(Wl1,  Wl1p,  CH, COUT);

    // ---- layer 0: GCN(IN->H) + BN + ReLU + residual projection (fully fused epilogue) ----
    gemm_bf16_wmma<CH><<<gemmGrid, wblk, 0, stream>>>(xh, W0p,   nullptr, hlin, NNODES, CIN);
    gemm_bf16_wmma<CH><<<gemmGrid, wblk, 0, stream>>>(xh, Wresp, nullptr, res,  NNODES, CIN);
    fill_f32<<<cdiv(NH, 256), blk256, 0, stream>>>(agg, 0.0f, NH);
    edge_aggregate<<<cdiv(NEDGES * 32, 256), blk256, 0, stream>>>(hlin, src, dst, dinv, agg, NEDGES);
    combine_bn_relu<true><<<cdiv(NH, 256), blk256, 0, stream>>>(agg, hlin, dinv, b0, g0, be0,
                                                                res, bres, h, hh, NH);

    // ---- layer 1 ----
    gemm_bf16_wmma<CH><<<gemmGrid, wblk, 0, stream>>>(hh, W1p, nullptr, hlin, NNODES, CH);
    fill_f32<<<cdiv(NH, 256), blk256, 0, stream>>>(agg, 0.0f, NH);
    edge_aggregate<<<cdiv(NEDGES * 32, 256), blk256, 0, stream>>>(hlin, src, dst, dinv, agg, NEDGES);
    combine_bn_relu<false><<<cdiv(NH, 256), blk256, 0, stream>>>(agg, hlin, dinv, b1, g1, be1,
                                                                 nullptr, nullptr, h, hh, NH);

    // ---- layer 2 ----
    gemm_bf16_wmma<CH><<<gemmGrid, wblk, 0, stream>>>(hh, W2p, nullptr, hlin, NNODES, CH);
    fill_f32<<<cdiv(NH, 256), blk256, 0, stream>>>(agg, 0.0f, NH);
    edge_aggregate<<<cdiv(NEDGES * 32, 256), blk256, 0, stream>>>(hlin, src, dst, dinv, agg, NEDGES);
    combine_bn_relu<false><<<cdiv(NH, 256), blk256, 0, stream>>>(agg, hlin, dinv, b2, g2, be2,
                                                                 nullptr, nullptr, h, hh, NH);

    // ---- head MLP: relu(h@Wl0 + bl0) @ Wl1 + bl1 -> out ----
    gemm_bf16_wmma<CH><<<gemmGrid, wblk, 0, stream>>>(hh, Wl0p, nullptr, hlin, NNODES, CH);
    relu_bias_cast<<<cdiv(NH, 256), blk256, 0, stream>>>(hlin, bl0, hh, NH);
    gemm_bf16_wmma<COUT><<<gemmGrid, wblk, 0, stream>>>(hh, Wl1p, bl1, out, NNODES, CH);

    // ---- global mean pool of h over graphs ----
    fill_f32<<<cdiv(NB * CH, 256), blk256, 0, stream>>>(pooled, 0.0f, NB * CH);
    fill_f32<<<1, blk256, 0, stream>>>(cnts, 0.0f, NB);
    count_nodes<<<cdiv(NNODES, 256), blk256, 0, stream>>>(batch, cnts, NNODES);
    pool_sum<<<cdiv(NNODES * 32, 256), blk256, 0, stream>>>(h, batch, pooled, NNODES);
    pool_div<<<cdiv(NB * CH, 256), blk256, 0, stream>>>(pooled, cnts, NB * CH);
}
